// PKMLinear_27874337751162
// MI455X (gfx1250) — matrix operand order repaired and checked by reading, not compiled, not run on hardware
//
#include <hip/hip_runtime.h>
#include <hip/hip_bf16.h>

typedef __attribute__((ext_vector_type(16))) __bf16 v16bf;
typedef __attribute__((ext_vector_type(8)))  float  v8f;

#define D_IN   2048
#define NCOL   8000     // HEADS * 2 * PKM_BASE
#define NTOK   8192
#define PKM    1000
#define HEADS  4
#define KH     8        // k_head
#define KOUT   32
#define KC     128      // K per staged chunk
#define NG     64       // N columns per group

// ws layout (bf16 elements), all regions 16B aligned:
#define W_ELEMS  ((size_t)NCOL * D_IN)   // 16,384,000
#define X_ELEMS  ((size_t)NTOK * D_IN)   // 16,777,216
#define WS_NEED  ((2 * W_ELEMS + 2 * X_ELEMS) * 2)   // bytes = 132,644,864

// ---------------------------------------------------------------------------
// Pre-pass: split fp32 -> bf16 hi/lo, stored in WMMA fragment-swizzled order.
// B (W) swizzle:  block=[ntile][kstep], lane = ((kw>>3)&1)*16 + (n&15),
//                 e = (kw&7) | (((kw>>4)&1)<<3)
// A (x) swizzle:  block=[mtile][kstep], lane = ((kw>>3)&1)*16 + (m&15),
//                 e = (kw&7) | (((kw>>4)&1)<<3)
// ---------------------------------------------------------------------------
__global__ __launch_bounds__(256)
void swizzle_w_kernel(const float* __restrict__ W,
                      __bf16* __restrict__ whi, __bf16* __restrict__ wlo)
{
    const int gid = blockIdx.x * 256 + threadIdx.x;      // 8000*128 threads
    const int row = gid >> 7;
    const int kb  = (gid & 127) * 16;
    const float* wp = W + (size_t)row * D_IN + kb;
    float4 f0 = *(const float4*)(wp + 0);
    float4 f1 = *(const float4*)(wp + 4);
    float4 f2 = *(const float4*)(wp + 8);
    float4 f3 = *(const float4*)(wp + 12);
    float v[16] = {f0.x,f0.y,f0.z,f0.w, f1.x,f1.y,f1.z,f1.w,
                   f2.x,f2.y,f2.z,f2.w, f3.x,f3.y,f3.z,f3.w};
    const int ct = row >> 4, ncol = row & 15;
#pragma unroll
    for (int u = 0; u < 16; ++u) {
        const int k = kb + u, kstep = k >> 5, kw = k & 31;
        const int half = (kw >> 3) & 1, e = (kw & 7) | (((kw >> 4) & 1) << 3);
        const size_t dst = ((size_t)(ct * 64 + kstep)) * 512
                         + (half * 16 + ncol) * 16 + e;
        const __bf16 h = (__bf16)v[u];
        whi[dst] = h;
        wlo[dst] = (__bf16)(v[u] - (float)h);
    }
}

__global__ __launch_bounds__(256)
void swizzle_x_kernel(const float* __restrict__ x,
                      __bf16* __restrict__ xhi, __bf16* __restrict__ xlo)
{
    const int gid = blockIdx.x * 256 + threadIdx.x;      // 8192*128 threads
    const int row = gid >> 7;
    const int kb  = (gid & 127) * 16;
    const float* xp = x + (size_t)row * D_IN + kb;
    float4 f0 = *(const float4*)(xp + 0);
    float4 f1 = *(const float4*)(xp + 4);
    float4 f2 = *(const float4*)(xp + 8);
    float4 f3 = *(const float4*)(xp + 12);
    float v[16] = {f0.x,f0.y,f0.z,f0.w, f1.x,f1.y,f1.z,f1.w,
                   f2.x,f2.y,f2.z,f2.w, f3.x,f3.y,f3.z,f3.w};
    const int mt = row >> 4, mrow = row & 15;
#pragma unroll
    for (int u = 0; u < 16; ++u) {
        const int k = kb + u, kstep = k >> 5, kw = k & 31;
        const int half = (kw >> 3) & 1;
        const int e    = (kw & 7) | (((kw >> 4) & 1) << 3);
        const size_t dst = ((size_t)(mt * 64 + kstep)) * 512
                         + (half * 16 + mrow) * 16 + e;
        const __bf16 h = (__bf16)v[u];
        xhi[dst] = h;
        xlo[dst] = (__bf16)(v[u] - (float)h);
    }
}

// ---------------------------------------------------------------------------
// Fast fused kernel: M=64 tokens/block, 512 threads (16 waves, 4x4 WMMA tiles)
// Double-buffered async-to-LDS staging pipelined one chunk ahead of WMMA.
// ---------------------------------------------------------------------------
#define STG_BUF_BYTES 65536           // per buffer: Ahi|Alo|Bhi|Blo, 16KB each
#define NITER (125 * 16)              // flattened (grp, kc) iterations

__global__ __launch_bounds__(512)
void pkm_fast_kernel(const __bf16* __restrict__ xhi, const __bf16* __restrict__ xlo,
                     const __bf16* __restrict__ whi, const __bf16* __restrict__ wlo,
                     const float* __restrict__ b_lin,
                     float* __restrict__ out_w, int* __restrict__ out_i)
{
    // 128 KB double-buffered staging (disjoint from tail region so the async
    // pipeline can run across group boundaries).
    __shared__ __align__(32) __bf16 stg[2][32768];
    __shared__ __align__(32) union {
        struct { float proj[64 * 65]; } p;          // padded: no bank conflicts
        struct {
            float lw[64 * 64]; int li[64 * 64];
            float cw[64 * 32]; int ci[64 * 32];
        } f;                                        // 48 KB
    } sm;

    const int tid  = threadIdx.x;
    const int lane = tid & 31;
    const int wv   = tid >> 5;      // 16 waves
    const int wr   = wv >> 2;       // tile row 0..3
    const int wc   = wv & 3;        // tile col 0..3
    const int mbase = blockIdx.x * 64;
    const int mt0   = mbase >> 4;   // absolute 16-token tile index base

    // LDS byte offset of staging base (low 32 bits of generic ptr = LDS offset)
    const uint32_t stg0 = (uint32_t)(size_t)&stg[0][0];

    // Per-thread staging slots: 1024 x 16B per region, 2 slots per thread.
    // slot -> block = slot>>6 (= ks*4 + r|c), elem offset = (slot&63)*8
    int      s_rc[2], s_ks[2];
    uint32_t s_lds[2];
    size_t   s_w8[2];
#pragma unroll
    for (int s = 0; s < 2; ++s) {
        const int slot = tid + s * 512;
        const int blk  = slot >> 6;
        s_ks[s]  = blk >> 2;
        s_rc[s]  = blk & 3;
        s_w8[s]  = (size_t)((slot & 63) * 8);
        s_lds[s] = (uint32_t)((blk * 512 + (slot & 63) * 8) * 2);
    }

    // Issue the 8 async copies for one chunk into staging buffer `b`.
    auto issue_chunk = [&](int ct0n, int kc4n, int b) {
        const uint32_t base = stg0 + (uint32_t)b * STG_BUF_BYTES;
#pragma unroll
        for (int s = 0; s < 2; ++s) {
            const size_t ga = ((size_t)((mt0 + s_rc[s]) * 64 + kc4n + s_ks[s])) * 512 + s_w8[s];
            const size_t gb = ((size_t)((ct0n + s_rc[s]) * 64 + kc4n + s_ks[s])) * 512 + s_w8[s];
            asm volatile("global_load_async_to_lds_b128 %0, %1, off"
                         :: "v"(base + 0     + s_lds[s]), "v"(xhi + ga) : "memory");
            asm volatile("global_load_async_to_lds_b128 %0, %1, off"
                         :: "v"(base + 16384 + s_lds[s]), "v"(xlo + ga) : "memory");
            asm volatile("global_load_async_to_lds_b128 %0, %1, off"
                         :: "v"(base + 32768 + s_lds[s]), "v"(whi + gb) : "memory");
            asm volatile("global_load_async_to_lds_b128 %0, %1, off"
                         :: "v"(base + 49152 + s_lds[s]), "v"(wlo + gb) : "memory");
        }
    };

    // Running top-8 of one (token, segment) per thread: 64 tok x 8 seg = 512
    const int myTok = tid >> 3;
    const int mySeg = tid & 7;
    float lw[KH]; int li[KH];
#pragma unroll
    for (int q = 0; q < KH; ++q) { lw[q] = -__builtin_inff(); li[q] = 0; }

    // prologue: prefetch first chunk
    issue_chunk(0, 0, 0);

    for (int grp = 0; grp < NCOL / NG; ++grp) {
        const int n0 = grp * NG;
        v8f acc = {0.f,0.f,0.f,0.f,0.f,0.f,0.f,0.f};

        for (int kc = 0; kc < D_IN / KC; ++kc) {
            const int b    = kc & 1;
            const int iter = grp * 16 + kc;

            // ---- pipeline: issue next chunk, then wait for current only
            if (iter + 1 < NITER) {
                const int ni = iter + 1;
                issue_chunk((ni >> 4) * 4, (ni & 15) * 4, ni & 1);
                // async loads complete in order: <=8 outstanding means the
                // current chunk's 8 copies have landed.
                asm volatile("s_wait_asynccnt 0x8" ::: "memory");
            } else {
                asm volatile("s_wait_asynccnt 0x0" ::: "memory");
            }
            __syncthreads();

            // ---- compute: 4 K-steps x 3 WMMAs (bf16x3 fp32-accurate split)
            const __bf16* Ah = &stg[b][0];
            const __bf16* Al = Ah + 8192;
            const __bf16* Bh = Ah + 16384;
            const __bf16* Bl = Ah + 24576;
#pragma unroll
            for (int ks = 0; ks < KC / 32; ++ks) {
                const v16bf ah = *(const v16bf*)&Ah[((ks * 4 + wr) * 32 + lane) * 16];
                const v16bf al = *(const v16bf*)&Al[((ks * 4 + wr) * 32 + lane) * 16];
                const v16bf bh = *(const v16bf*)&Bh[((ks * 4 + wc) * 32 + lane) * 16];
                const v16bf bl = *(const v16bf*)&Bl[((ks * 4 + wc) * 32 + lane) * 16];
                acc = __builtin_amdgcn_wmma_f32_16x16x32_bf16(false, al, false, bh,
                                                              (short)0, acc, false, false);
                acc = __builtin_amdgcn_wmma_f32_16x16x32_bf16(false, ah, false, bl,
                                                              (short)0, acc, false, false);
                acc = __builtin_amdgcn_wmma_f32_16x16x32_bf16(false, ah, false, bh,
                                                              (short)0, acc, false, false);
            }
            __syncthreads();
        }

        // ---- dump C tile (+ b_lin) into padded proj tile
        {
            const int colg  = wc * 16 + (lane & 15);
            const float bl  = b_lin[n0 + colg];
            const int mrow0 = wr * 16 + (lane >> 4) * 8;
#pragma unroll
            for (int j = 0; j < 8; ++j)
                sm.p.proj[(mrow0 + j) * 65 + colg] = acc[j] + bl;
        }
        __syncthreads();

        // ---- streaming stable top-8 (scan ascending idx, strict >)
        {
            int lo = mySeg * PKM;       if (lo < n0)      lo = n0;
            int hi = mySeg * PKM + PKM; if (hi > n0 + NG) hi = n0 + NG;
            for (int col = lo; col < hi; ++col) {
                const float v = sm.p.proj[myTok * 65 + (col - n0)];
                if (v > lw[KH - 1]) {
                    float vv = v; int ii = col - mySeg * PKM;
#pragma unroll
                    for (int q = 0; q < KH; ++q) {
                        if (vv > lw[q]) {
                            float tw = lw[q]; int ti = li[q];
                            lw[q] = vv; li[q] = ii; vv = tw; ii = ti;
                        }
                    }
                }
            }
        }
        __syncthreads();
    }

    // ================= tail: per-head combine + final sort =================
#pragma unroll
    for (int q = 0; q < KH; ++q) {
        sm.f.lw[myTok * 64 + mySeg * 8 + q] = lw[q];
        sm.f.li[myTok * 64 + mySeg * 8 + q] = li[q];
    }
    __syncthreads();

    if (tid < 64 * HEADS) {          // one thread per (token, head)
        const int t = tid >> 2, h = tid & 3;
        float cw[KH]; int ci[KH];
#pragma unroll
        for (int q = 0; q < KH; ++q) { cw[q] = -__builtin_inff(); ci[q] = 0; }
        const int b1 = t * 64 + (2 * h)     * 8;   // x1 list
        const int b2 = t * 64 + (2 * h + 1) * 8;   // x2 list
#pragma unroll
        for (int i = 0; i < KH; ++i) {
            const float w1 = sm.f.lw[b1 + i];
            const int   i1 = sm.f.li[b1 + i];
#pragma unroll
            for (int j = 0; j < KH; ++j) {
                float s = w1 + sm.f.lw[b2 + j];
                if (s < 0.f) s = 0.f;                       // relu
                const int code = i1 * PKM + sm.f.li[b2 + j];
                if (s > cw[KH - 1]) {
                    float vv = s; int ii = code;
#pragma unroll
                    for (int q = 0; q < KH; ++q) {
                        if (vv > cw[q]) {
                            float tw = cw[q]; int ti = ci[q];
                            cw[q] = vv; ci[q] = ii; vv = tw; ii = ti;
                        }
                    }
                }
            }
        }
#pragma unroll
        for (int q = 0; q < KH; ++q) {
            sm.f.cw[t * 32 + h * 8 + q] = cw[q];
            sm.f.ci[t * 32 + h * 8 + q] = ci[q];
        }
    }
    __syncthreads();

    if (tid < 64) {                  // one thread per token: stable sort-32
        const int t = tid;
        float* ow = out_w + (size_t)(mbase + t) * KOUT;
        int*   oi = out_i + (size_t)(mbase + t) * KOUT;
        for (int p = 0; p < KOUT; ++p) {
            float best = -__builtin_inff(); int bj = 0;
            for (int j = 0; j < 32; ++j) {
                const float v = sm.f.cw[t * 32 + j];
                if (v > best) { best = v; bj = j; }
            }
            ow[p] = best;
            oi[p] = sm.f.ci[t * 32 + bj];
            sm.f.cw[t * 32 + bj] = -__builtin_inff();
        }
    }
}

// ---------------------------------------------------------------------------
// Fallback fused kernel (no workspace): M=32/block, converts in-kernel.
// ---------------------------------------------------------------------------
__global__ __launch_bounds__(256)
void pkm_fused_kernel(const float* __restrict__ x,
                      const float* __restrict__ W,
                      const float* __restrict__ b_lin,
                      float* __restrict__ out_w,
                      int*   __restrict__ out_i)
{
    __shared__ __align__(32) union {
        struct {
            __bf16 Ahi[4096]; __bf16 Alo[4096];
            __bf16 Bhi[8192]; __bf16 Blo[8192];
        } g;
        struct { float proj[32 * 65]; } p;
        struct {
            float lw[32 * 64]; int li[32 * 64];
            float cw[32 * 32]; int ci[32 * 32];
        } f;
    } sm;

    const int tid  = threadIdx.x;
    const int lane = tid & 31;
    const int wv   = tid >> 5;
    const int wr   = wv >> 2;
    const int wc   = wv & 3;
    const int mbase = blockIdx.x * 32;

    const int myTok = tid >> 3;
    const int mySeg = tid & 7;
    float lw[KH]; int li[KH];
#pragma unroll
    for (int q = 0; q < KH; ++q) { lw[q] = -__builtin_inff(); li[q] = 0; }

    for (int grp = 0; grp < NCOL / NG; ++grp) {
        const int n0 = grp * NG;
        v8f acc = {0.f,0.f,0.f,0.f,0.f,0.f,0.f,0.f};

        for (int kc = 0; kc < D_IN / KC; ++kc) {
            const int k0 = kc * KC;
            {
                const int m  = tid >> 3;
                const int kb = (tid & 7) * 16;
                const float* xp = x + (size_t)(mbase + m) * D_IN + k0 + kb;
                float4 f0 = *(const float4*)(xp + 0);
                float4 f1 = *(const float4*)(xp + 4);
                float4 f2 = *(const float4*)(xp + 8);
                float4 f3 = *(const float4*)(xp + 12);
                float v[16] = {f0.x,f0.y,f0.z,f0.w, f1.x,f1.y,f1.z,f1.w,
                               f2.x,f2.y,f2.z,f2.w, f3.x,f3.y,f3.z,f3.w};
                const int r = m >> 4, mrow = m & 15;
#pragma unroll
                for (int u = 0; u < 16; ++u) {
                    const int kk   = kb + u;
                    const int ks   = kk >> 5;
                    const int kw   = kk & 31;
                    const int half = (kw >> 3) & 1;
                    const int e    = (kw & 7) | (((kw >> 4) & 1) << 3);
                    const int idx  = (((ks * 2 + r) * 32) + (half * 16 + mrow)) * 16 + e;
                    const __bf16 h = (__bf16)v[u];
                    sm.g.Ahi[idx] = h;
                    sm.g.Alo[idx] = (__bf16)(v[u] - (float)h);
                }
            }
            {
                const int n  = tid >> 2;
                const int kb = (tid & 3) * 32;
                const float* wp = W + (size_t)(n0 + n) * D_IN + k0 + kb;
                const int c = n >> 4, ncol = n & 15;
#pragma unroll
                for (int u4 = 0; u4 < 8; ++u4) {
                    float4 fq = *(const float4*)(wp + u4 * 4);
                    float vv[4] = {fq.x, fq.y, fq.z, fq.w};
#pragma unroll
                    for (int t = 0; t < 4; ++t) {
                        const int kk   = kb + u4 * 4 + t;
                        const int ks   = kk >> 5;
                        const int kw   = kk & 31;
                        const int half = (kw >> 3) & 1;
                        const int e    = (kw & 7) | (((kw >> 4) & 1) << 3);
                        const int idx  = (((ks * 4 + c) * 32) + (half * 16 + ncol)) * 16 + e;
                        const __bf16 h = (__bf16)vv[t];
                        sm.g.Bhi[idx] = h;
                        sm.g.Blo[idx] = (__bf16)(vv[t] - (float)h);
                    }
                }
            }
            __syncthreads();
#pragma unroll
            for (int ks = 0; ks < KC / 32; ++ks) {
                const v16bf ah = *(const v16bf*)&sm.g.Ahi[((ks * 2 + wr) * 32 + lane) * 16];
                const v16bf al = *(const v16bf*)&sm.g.Alo[((ks * 2 + wr) * 32 + lane) * 16];
                const v16bf bh = *(const v16bf*)&sm.g.Bhi[((ks * 4 + wc) * 32 + lane) * 16];
                const v16bf bl = *(const v16bf*)&sm.g.Blo[((ks * 4 + wc) * 32 + lane) * 16];
                acc = __builtin_amdgcn_wmma_f32_16x16x32_bf16(false, al, false, bh,
                                                              (short)0, acc, false, false);
                acc = __builtin_amdgcn_wmma_f32_16x16x32_bf16(false, ah, false, bl,
                                                              (short)0, acc, false, false);
                acc = __builtin_amdgcn_wmma_f32_16x16x32_bf16(false, ah, false, bh,
                                                              (short)0, acc, false, false);
            }
            __syncthreads();
        }
        {
            const int colg  = wc * 16 + (lane & 15);
            const float bl  = b_lin[n0 + colg];
            const int mrow0 = wr * 16 + (lane >> 4) * 8;
#pragma unroll
            for (int j = 0; j < 8; ++j)
                sm.p.proj[(mrow0 + j) * 65 + colg] = acc[j] + bl;
        }
        __syncthreads();
        {
            int lo = mySeg * PKM;       if (lo < n0)      lo = n0;
            int hi = mySeg * PKM + PKM; if (hi > n0 + NG) hi = n0 + NG;
            for (int col = lo; col < hi; ++col) {
                const float v = sm.p.proj[myTok * 65 + (col - n0)];
                if (v > lw[KH - 1]) {
                    float vv = v; int ii = col - mySeg * PKM;
#pragma unroll
                    for (int q = 0; q < KH; ++q) {
                        if (vv > lw[q]) {
                            float tw = lw[q]; int ti = li[q];
                            lw[q] = vv; li[q] = ii; vv = tw; ii = ti;
                        }
                    }
                }
            }
        }
        __syncthreads();
    }

#pragma unroll
    for (int q = 0; q < KH; ++q) {
        sm.f.lw[myTok * 64 + mySeg * 8 + q] = lw[q];
        sm.f.li[myTok * 64 + mySeg * 8 + q] = li[q];
    }
    __syncthreads();

    if (tid < 32 * HEADS) {
        const int t = tid >> 2, h = tid & 3;
        float cw[KH]; int ci[KH];
#pragma unroll
        for (int q = 0; q < KH; ++q) { cw[q] = -__builtin_inff(); ci[q] = 0; }
        const int b1 = t * 64 + (2 * h)     * 8;
        const int b2 = t * 64 + (2 * h + 1) * 8;
#pragma unroll
        for (int i = 0; i < KH; ++i) {
            const float w1 = sm.f.lw[b1 + i];
            const int   i1 = sm.f.li[b1 + i];
#pragma unroll
            for (int j = 0; j < KH; ++j) {
                float s = w1 + sm.f.lw[b2 + j];
                if (s < 0.f) s = 0.f;
                const int code = i1 * PKM + sm.f.li[b2 + j];
                if (s > cw[KH - 1]) {
                    float vv = s; int ii = code;
#pragma unroll
                    for (int q = 0; q < KH; ++q) {
                        if (vv > cw[q]) {
                            float tw = cw[q]; int ti = ci[q];
                            cw[q] = vv; ci[q] = ii; vv = tw; ii = ti;
                        }
                    }
                }
            }
        }
#pragma unroll
        for (int q = 0; q < KH; ++q) {
            sm.f.cw[t * 32 + h * 8 + q] = cw[q];
            sm.f.ci[t * 32 + h * 8 + q] = ci[q];
        }
    }
    __syncthreads();

    if (tid < 32) {
        const int t = tid;
        float* ow = out_w + (size_t)(mbase + t) * KOUT;
        int*   oi = out_i + (size_t)(mbase + t) * KOUT;
        for (int p = 0; p < KOUT; ++p) {
            float best = -__builtin_inff(); int bj = 0;
            for (int j = 0; j < 32; ++j) {
                const float v = sm.f.cw[t * 32 + j];
                if (v > best) { best = v; bj = j; }
            }
            ow[p] = best;
            oi[p] = sm.f.ci[t * 32 + bj];
            sm.f.cw[t * 32 + bj] = -__builtin_inff();
        }
    }
}

extern "C" void kernel_launch(void* const* d_in, const int* in_sizes, int n_in,
                              void* d_out, int out_size, void* d_ws, size_t ws_size,
                              hipStream_t stream) {
    (void)in_sizes; (void)n_in; (void)out_size;
    const float* x     = (const float*)d_in[0];
    const float* W     = (const float*)d_in[1];
    const float* b_lin = (const float*)d_in[2];
    // d_in[3] (bias) and d_in[4] (k) are dead: i1*1000+i2 < NUM_LATENTS always.
    float* out_w = (float*)d_out;                       // (8192, 32) f32
    int*   out_i = (int*)d_out + (size_t)NTOK * KOUT;   // (8192, 32) i32

    if (ws_size >= WS_NEED) {
        __bf16* whi = (__bf16*)d_ws;
        __bf16* wlo = whi + W_ELEMS;
        __bf16* xhi = wlo + W_ELEMS;
        __bf16* xlo = xhi + X_ELEMS;
        swizzle_w_kernel<<<(NCOL * (D_IN / 16)) / 256, 256, 0, stream>>>(W, whi, wlo);
        swizzle_x_kernel<<<(NTOK * (D_IN / 16)) / 256, 256, 0, stream>>>(x, xhi, xlo);
        pkm_fast_kernel<<<NTOK / 64, 512, 0, stream>>>(xhi, xlo, whi, wlo,
                                                       b_lin, out_w, out_i);
    } else {
        pkm_fused_kernel<<<NTOK / 32, 256, 0, stream>>>(x, W, b_lin, out_w, out_i);
    }
}
